// DCInstDynKernelGenerator_66322884985166
// MI455X (gfx1250) — matrix operand order repaired and checked by reading, not compiled, not run on hardware
//
#include <hip/hip_runtime.h>

typedef __attribute__((ext_vector_type(16))) __bf16 v16bf;
typedef __attribute__((ext_vector_type(8)))  __bf16 v8bf;
typedef __attribute__((ext_vector_type(8)))  float  v8f;
typedef __attribute__((ext_vector_type(4)))  unsigned int v4u;
typedef __attribute__((ext_vector_type(8)))  int v8i;
typedef __attribute__((ext_vector_type(4)))  int v4i;

#define HH 96
#define WW 96
#define NPIX (HH * WW)          // 9216
#define NIMG 16                 // b * n_f
#define CDCN 64

// ---------------------------------------------------------------------------
// small helper kernels
// ---------------------------------------------------------------------------

// transpose-convert weights (O x K f32) -> (K x 64 bf16), zero-pad o >= Cout
__global__ void f2bfT_kernel(const float* __restrict__ src, __bf16* __restrict__ dst,
                             int K, int Cout) {
    int i = blockIdx.x * 256 + threadIdx.x;
    if (i < K * 64) {
        int k = i >> 6, o = i & 63;
        dst[i] = (o < Cout) ? (__bf16)src[(long)o * K + k] : (__bf16)0.0f;
    }
}

// concat(off_w(18,576), mod_w(9,576)) -> transposed (576 x 64 bf16), pad to 64
__global__ void f2bfT27_kernel(const float* __restrict__ off_w,
                               const float* __restrict__ mod_w,
                               __bf16* __restrict__ dst) {
    int i = blockIdx.x * 256 + threadIdx.x;
    if (i < 576 * 64) {
        int k = i >> 6, o = i & 63;
        float v = 0.0f;
        if (o < 18)      v = off_w[(long)o * 576 + k];
        else if (o < 27) v = mod_w[(long)(o - 18) * 576 + k];
        dst[i] = (__bf16)v;
    }
}

__global__ void prep_bn_kernel(const float* __restrict__ conv_b,
                               const float* __restrict__ gamma,
                               const float* __restrict__ beta,
                               const float* __restrict__ mean,
                               const float* __restrict__ var,
                               float* __restrict__ scale, float* __restrict__ bias) {
    int o = threadIdx.x;
    if (o < CDCN) {
        float s = gamma[o] * rsqrtf(var[o] + 1e-5f);
        scale[o] = s;
        bias[o]  = beta[o] + (conv_b[o] - mean[o]) * s;
    }
}

__global__ void prep27_kernel(const float* __restrict__ off_b,
                              const float* __restrict__ mod_b,
                              float* __restrict__ scale, float* __restrict__ bias) {
    int o = threadIdx.x;
    if (o < 27) {
        scale[o] = 1.0f;
        bias[o]  = (o < 18) ? off_b[o] : mod_b[o - 18];
    }
}

// ---------------------------------------------------------------------------
// Tensor Data Mover: load one contiguous (32 x 64 bf16) weight tile into LDS,
// padding 4 dwords after every 32 dwords so LDS layout is [32][72] bf16
// (bank-conflict-avoiding row stride of 144B) — the DMA does the swizzle.
//   D# group0: count=1, lds_addr, global_addr, type=2
//   D# group1: data_size=1 (2B), pad_enable, pad_interval=4 (32 dw),
//              pad_amount=3 (4 dw), 1-D tensor/tile of 2048 elements
// ---------------------------------------------------------------------------
__device__ __forceinline__ void tdm_load_b_tile(const __bf16* gsrc, unsigned lds_off) {
    unsigned long long ga = (unsigned long long)(uintptr_t)gsrc;
    v4u g0;
    g0[0] = 1u;                                        // count=1 (valid user D#)
    g0[1] = lds_off;                                   // lds_addr (bytes)
    g0[2] = (unsigned)(ga & 0xffffffffu);              // global_addr[31:0]
    g0[3] = (unsigned)((ga >> 32) & 0x1ffffffu) | (2u << 30);  // [56:32] | type=2
    v8i g1;
    g1[0] = (int)((1u << 16) | (1u << 20) | (4u << 22) | (3u << 25));
    g1[1] = (int)(2048u << 16);       // tensor_dim0[15:0] @ bits[63:48]
    g1[2] = (int)(1u << 16);          // tensor_dim0[31:16]=0, tensor_dim1=1
    g1[3] = (int)(2048u << 16);       // tensor_dim1 hi=0, tile_dim0=2048
    g1[4] = 0;                        // tile_dim1/2 unused
    g1[5] = 2048;                     // tensor_dim0_stride[31:0]
    g1[6] = 0;
    g1[7] = 0;
    v4i z4 = {0, 0, 0, 0};
#if __clang_major__ >= 23
    v8i z8 = {0, 0, 0, 0, 0, 0, 0, 0};
    __builtin_amdgcn_tensor_load_to_lds(g0, g1, z4, z4, z8, 0);
#else
    __builtin_amdgcn_tensor_load_to_lds(g0, g1, z4, z4, 0);
#endif
}

// ---------------------------------------------------------------------------
// fragment helpers (CDNA5 WMMA bf16 layouts, wave32)
// ---------------------------------------------------------------------------
__device__ __forceinline__ v16bf load_a_frag(const __bf16* rowptr, int lane) {
    int kb = (lane < 16) ? 0 : 8;
    v8bf lo = *(const v8bf*)(rowptr + kb);
    v8bf hi = *(const v8bf*)(rowptr + kb + 16);
    v16bf a;
#pragma unroll
    for (int i = 0; i < 8; ++i) { a[i] = lo[i]; a[8 + i] = hi[i]; }
    return a;
}

__device__ __forceinline__ v16bf load_b_frag(const __bf16* krowptr, int nbase) {
    v16bf b;
#pragma unroll
    for (int i = 0; i < 8; ++i) { b[i] = krowptr[i * 72 + nbase]; b[8 + i] = krowptr[(16 + i) * 72 + nbase]; }
    return b;
}

// ---------------------------------------------------------------------------
// implicit-im2col conv3x3 (stride 1, pad 1) as WMMA GEMM; weights pre-
// transposed to [k][64] bf16 so the B tile is a contiguous TDM transfer.
// grid: (NPIX/64, NIMG), block 128 (4 waves); wave computes 32px x 32ch.
// ---------------------------------------------------------------------------
template <int CIN, int COUT>
__global__ void conv3x3_wmma_kernel(const float* __restrict__ x,
                                    const __bf16* __restrict__ wbT,
                                    const float* __restrict__ scale,
                                    const float* __restrict__ bias,
                                    float* __restrict__ out, int relu) {
    __shared__ __align__(16) __bf16 Alds[64][40];
    __shared__ __align__(16) __bf16 Blds[32][72];

    const int tid  = threadIdx.x;
    const int lane = tid & 31;
    const int wid  = tid >> 5;
    const int msub = (wid & 1) * 32;
    const int nsub = (wid >> 1) * 32;
    const int n     = blockIdx.y;
    const int tileM = blockIdx.x * 64;
    const int K     = CIN * 9;
    const unsigned blds_off = (unsigned)(uintptr_t)(&Blds[0][0]);

    v8f acc[2][2] = {{{}, {}}, {{}, {}}};

    for (int kk = 0; kk < K; kk += 32) {
        // async DMA of the weight tile (wave 0 only; EXEC-independent op)
        if (wid == 0) tdm_load_b_tile(wbT + (size_t)kk * 64, blds_off);

        // stage A tile: this thread's K-column is loop-invariant (128 % 32 == 0)
        {
            int k = kk + lane;
            int c = k / 9, t = k - c * 9;
            int dy = t / 3 - 1;
            int dx = t - (t / 3) * 3 - 1;
            const float* xc = x + ((long)n * CIN + c) * NPIX;
            int p0 = tileM + wid;
            int ph = p0 / WW, pw = p0 - ph * WW;
            __bf16* acol = &Alds[wid][lane];
#pragma unroll
            for (int i = 0; i < 16; ++i) {
                int hy = ph + dy, hx = pw + dx;
                float v = 0.0f;
                if ((unsigned)hy < (unsigned)HH && (unsigned)hx < (unsigned)WW)
                    v = xc[hy * WW + hx];
                acol[(size_t)i * 4 * 40] = (__bf16)v;      // row = wid + 4*i
                pw += 4;
                if (pw >= WW) { pw -= WW; ph += 1; }
            }
            if (kk + 32 < K) {  // warm L2 for the next K-step's source line
                int c2 = (k + 32) / 9;
                __builtin_prefetch(x + ((long)n * CIN + c2) * NPIX + p0, 0, 1);
            }
        }
        if (wid == 0) __builtin_amdgcn_s_wait_tensorcnt(0);
        __syncthreads();

        v16bf afrag[2], bfrag[2];
#pragma unroll
        for (int mi = 0; mi < 2; ++mi)
            afrag[mi] = load_a_frag(&Alds[msub + mi * 16 + (lane & 15)][0], lane);
#pragma unroll
        for (int ni = 0; ni < 2; ++ni)
            bfrag[ni] = load_b_frag(&Blds[(lane >> 4) * 8][0], nsub + ni * 16 + (lane & 15));
#pragma unroll
        for (int mi = 0; mi < 2; ++mi)
#pragma unroll
            for (int ni = 0; ni < 2; ++ni)
                acc[mi][ni] = __builtin_amdgcn_wmma_f32_16x16x32_bf16(
                    false, afrag[mi], false, bfrag[ni], (short)0, acc[mi][ni],
                    false, false);
        __syncthreads();
    }

#pragma unroll
    for (int mi = 0; mi < 2; ++mi)
#pragma unroll
        for (int ni = 0; ni < 2; ++ni) {
            int o = nsub + ni * 16 + (lane & 15);
            if (o >= COUT) continue;
            float s = scale[o], bb = bias[o];
            int pbase = tileM + msub + mi * 16 + (lane >> 4) * 8;
            long obase = ((long)n * COUT + o) * NPIX;
#pragma unroll
            for (int i = 0; i < 8; ++i) {
                float v = fmaf(acc[mi][ni][i], s, bb);
                if (relu) v = fmaxf(v, 0.0f);
                out[obase + pbase + i] = v;
            }
        }
}

// ---------------------------------------------------------------------------
// DCNv2: gather-built A tile (bilinear sample * modulation), TDM-loaded B.
// out27 channels: [dy0,dx0,...,dy8,dx8, mod0..mod8].  K = 576.
// ---------------------------------------------------------------------------
__global__ void dcn_wmma_kernel(const float* __restrict__ y1,
                                const float* __restrict__ out27,
                                const __bf16* __restrict__ wbdT,
                                const float* __restrict__ dcn_b,
                                float* __restrict__ y2) {
    __shared__ __align__(16) __bf16 Alds[64][40];
    __shared__ __align__(16) __bf16 Blds[32][72];
    __shared__ int   sy[576];
    __shared__ int   sx[576];
    __shared__ float swy[576];
    __shared__ float swx[576];
    __shared__ float smv[576];

    const int tid  = threadIdx.x;
    const int lane = tid & 31;
    const int wid  = tid >> 5;
    const int msub = (wid & 1) * 32;
    const int nsub = (wid >> 1) * 32;
    const int n     = blockIdx.y;
    const int tileM = blockIdx.x * 64;
    const int K     = CDCN * 9;   // 576
    const unsigned blds_off = (unsigned)(uintptr_t)(&Blds[0][0]);

    // stage 0: per-(pixel, tap) sampling parameters
    for (int s = tid; s < 576; s += 128) {
        int r = s / 9, t = s - r * 9;
        int p = tileM + r;
        int ph = p / WW, pw = p - ph * WW;
        long base = (long)n * 27 * NPIX + p;
        float dyv  = out27[base + (long)(2 * t) * NPIX];
        float dxv  = out27[base + (long)(2 * t + 1) * NPIX];
        float mraw = out27[base + (long)(18 + t) * NPIX];
        float m = 2.0f / (1.0f + __expf(-mraw));
        float py = dyv + (float)(t / 3 - 1) + (float)ph;
        float px = dxv + (float)(t - (t / 3) * 3 - 1) + (float)pw;
        float fy = floorf(py), fx = floorf(px);
        sy[s]  = (int)fy;
        sx[s]  = (int)fx;
        swy[s] = py - fy;
        swx[s] = px - fx;
        smv[s] = m;
    }
    __syncthreads();

    v8f acc[2][2] = {{{}, {}}, {{}, {}}};

    for (int kk = 0; kk < K; kk += 32) {
        if (wid == 0) tdm_load_b_tile(wbdT + (size_t)kk * 64, blds_off);

        {   // A tile: K-column invariant per thread; clamp + mask (no branches)
            int k = kk + lane;
            int c = k / 9, t = k - c * 9;
            const float* ch = y1 + ((long)n * CDCN + c) * NPIX;
            int s = wid * 9 + t;                 // row r = wid + 4*i
            __bf16* acol = &Alds[wid][lane];
#pragma unroll
            for (int i = 0; i < 16; ++i) {
                int y0 = sy[s], x0 = sx[s];
                float wy = swy[s], wx = swx[s], m = smv[s];
                int y0c = min(max(y0, 0), HH - 1), y1c = min(max(y0 + 1, 0), HH - 1);
                int x0c = min(max(x0, 0), WW - 1), x1c = min(max(x0 + 1, 0), WW - 1);
                float fy0 = ((unsigned)y0 < (unsigned)HH) ? 1.0f : 0.0f;
                float fy1 = ((unsigned)(y0 + 1) < (unsigned)HH) ? 1.0f : 0.0f;
                float fx0 = ((unsigned)x0 < (unsigned)WW) ? 1.0f : 0.0f;
                float fx1 = ((unsigned)(x0 + 1) < (unsigned)WW) ? 1.0f : 0.0f;
                float g00 = ch[y0c * WW + x0c] * (fy0 * fx0);
                float g01 = ch[y0c * WW + x1c] * (fy0 * fx1);
                float g10 = ch[y1c * WW + x0c] * (fy1 * fx0);
                float g11 = ch[y1c * WW + x1c] * (fy1 * fx1);
                float v = ((g00 * (1.0f - wy) + g10 * wy) * (1.0f - wx) +
                           (g01 * (1.0f - wy) + g11 * wy) * wx) * m;
                acol[(size_t)i * 4 * 40] = (__bf16)v;
                s += 36;
            }
        }
        if (wid == 0) __builtin_amdgcn_s_wait_tensorcnt(0);
        __syncthreads();

        v16bf afrag[2], bfrag[2];
#pragma unroll
        for (int mi = 0; mi < 2; ++mi)
            afrag[mi] = load_a_frag(&Alds[msub + mi * 16 + (lane & 15)][0], lane);
#pragma unroll
        for (int ni = 0; ni < 2; ++ni)
            bfrag[ni] = load_b_frag(&Blds[(lane >> 4) * 8][0], nsub + ni * 16 + (lane & 15));
#pragma unroll
        for (int mi = 0; mi < 2; ++mi)
#pragma unroll
            for (int ni = 0; ni < 2; ++ni)
                acc[mi][ni] = __builtin_amdgcn_wmma_f32_16x16x32_bf16(
                    false, afrag[mi], false, bfrag[ni], (short)0, acc[mi][ni],
                    false, false);
        __syncthreads();
    }

#pragma unroll
    for (int mi = 0; mi < 2; ++mi)
#pragma unroll
        for (int ni = 0; ni < 2; ++ni) {
            int o = nsub + ni * 16 + (lane & 15);
            float bb = dcn_b[o];
            int pbase = tileM + msub + mi * 16 + (lane >> 4) * 8;
            long obase = ((long)n * CDCN + o) * NPIX;
#pragma unroll
            for (int i = 0; i < 8; ++i)
                y2[obase + pbase + i] = acc[mi][ni][i] + bb;
        }
}

// ---------------------------------------------------------------------------
// mask-weighted pooling. 192->96 bilinear (align_corners=False, scale 2) is an
// exact 2x2 average. grid: (16 bf, 10 masks), 256 threads.
// ---------------------------------------------------------------------------
__global__ void pool_kernel(const float* __restrict__ y2,
                            const float* __restrict__ masks,
                            float* __restrict__ num, float* __restrict__ den) {
    __shared__ float mt[1024];
    __shared__ float red[256];
    const int bf = blockIdx.x, im = blockIdx.y, tid = threadIdx.x;
    const float* mk = masks + ((long)bf * 10 + im) * (192 * 192);
    const float* xc = y2 + (long)bf * CDCN * NPIX;
    const int c = tid & 63, pg = tid >> 6;
    float partial = 0.0f, dacc = 0.0f;

    for (int chunk = 0; chunk < NPIX; chunk += 1024) {
        for (int l = tid; l < 1024; l += 256) {
            int p = chunk + l;
            int ph = p / WW, pw = p - ph * WW;
            const float* rr = mk + (2 * ph) * 192 + 2 * pw;
            float m = 0.25f * (rr[0] + rr[1] + rr[192] + rr[193]);
            mt[l] = m;
            dacc += m;
        }
        __syncthreads();
        for (int l = pg; l < 1024; l += 4)
            partial += xc[(long)c * NPIX + chunk + l] * mt[l];
        __syncthreads();
    }

    red[tid] = partial;
    __syncthreads();
    if (tid < 64)
        num[((long)bf * 10 + im) * 64 + tid] =
            red[tid] + red[tid + 64] + red[tid + 128] + red[tid + 192];
    __syncthreads();
    red[tid] = dacc;
    __syncthreads();
    for (int s = 128; s > 0; s >>= 1) {
        if (tid < s) red[tid] += red[tid + s];
        __syncthreads();
    }
    if (tid == 0) den[bf * 10 + im] = red[0];
}

// ---------------------------------------------------------------------------
// FC head: feat = num/(den+eps); hid = relu(feat @ smooth^T + b);
// inc = hid @ inc_w^T (128), pix = hid @ pix_w^T (64). 160 rows.
// ---------------------------------------------------------------------------
__global__ void head_kernel(const float* __restrict__ num,
                            const float* __restrict__ den,
                            const float* __restrict__ smooth_w,
                            const float* __restrict__ smooth_b,
                            const float* __restrict__ inc_w,
                            const float* __restrict__ pix_w,
                            float* __restrict__ out) {
    __shared__ float feat[64];
    __shared__ float hid[64];
    const int row = blockIdx.x, tid = threadIdx.x;
    if (tid < 64) feat[tid] = num[(long)row * 64 + tid] / (den[row] + 1e-6f);
    __syncthreads();
    if (tid < 64) {
        float a = smooth_b[tid];
        for (int k = 0; k < 64; ++k) a += feat[k] * smooth_w[tid * 64 + k];
        hid[tid] = fmaxf(a, 0.0f);
    }
    __syncthreads();
    {
        float a = 0.0f;
        for (int k = 0; k < 64; ++k) a += hid[k] * inc_w[tid * 64 + k];
        out[(long)row * 128 + tid] = a;                 // inc_kernels
    }
    if (tid < 64) {
        float a = 0.0f;
        for (int k = 0; k < 64; ++k) a += hid[k] * pix_w[tid * 64 + k];
        out[20480 + (long)row * 64 + tid] = a;          // pix_kernels
    }
}

// ---------------------------------------------------------------------------
extern "C" void kernel_launch(void* const* d_in, const int* in_sizes, int n_in,
                              void* d_out, int out_size, void* d_ws, size_t ws_size,
                              hipStream_t stream) {
    const float* x        = (const float*)d_in[0];
    const float* masks    = (const float*)d_in[1];
    const float* conv1_w  = (const float*)d_in[2];
    const float* conv1_b  = (const float*)d_in[3];
    const float* bn_gamma = (const float*)d_in[4];
    const float* bn_beta  = (const float*)d_in[5];
    const float* bn_mean  = (const float*)d_in[6];
    const float* bn_var   = (const float*)d_in[7];
    const float* off_w    = (const float*)d_in[8];
    const float* off_b    = (const float*)d_in[9];
    const float* mod_w    = (const float*)d_in[10];
    const float* mod_b    = (const float*)d_in[11];
    const float* dcn_w    = (const float*)d_in[12];
    const float* dcn_b    = (const float*)d_in[13];
    const float* smooth_w = (const float*)d_in[14];
    const float* smooth_b = (const float*)d_in[15];
    const float* inc_w    = (const float*)d_in[16];
    const float* pix_w    = (const float*)d_in[17];

    // workspace carve-up (bytes, 256-aligned)
    char* ws = (char*)d_ws;
    size_t off = 0;
    auto carve = [&](size_t bytes) {
        char* p = ws + off;
        off = (off + bytes + 255) & ~(size_t)255;
        return p;
    };
    float*  y1      = (float*)carve((size_t)NIMG * CDCN * NPIX * 4);   // 37.7 MB
    float*  out27   = (float*)carve((size_t)NIMG * 27 * NPIX * 4);     // 15.9 MB
    float*  y2      = (float*)carve((size_t)NIMG * CDCN * NPIX * 4);   // 37.7 MB
    __bf16* wb1T    = (__bf16*)carve((size_t)2304 * 64 * 2);
    __bf16* wb2T    = (__bf16*)carve((size_t)576 * 64 * 2);
    __bf16* wbdT    = (__bf16*)carve((size_t)576 * 64 * 2);
    float*  scale1  = (float*)carve(64 * 4);
    float*  bias1   = (float*)carve(64 * 4);
    float*  scale27 = (float*)carve(32 * 4);
    float*  bias27  = (float*)carve(32 * 4);
    float*  numb    = (float*)carve(160 * 64 * 4);
    float*  denb    = (float*)carve(160 * 4);

    // weight transpose/convert + epilogue-constant prep
    f2bfT_kernel<<<(2304 * 64 + 255) / 256, 256, 0, stream>>>(conv1_w, wb1T, 2304, 64);
    f2bfT27_kernel<<<(576 * 64 + 255) / 256, 256, 0, stream>>>(off_w, mod_w, wb2T);
    f2bfT_kernel<<<(576 * 64 + 255) / 256, 256, 0, stream>>>(dcn_w, wbdT, 576, 64);
    prep_bn_kernel<<<1, 64, 0, stream>>>(conv1_b, bn_gamma, bn_beta, bn_mean, bn_var,
                                         scale1, bias1);
    prep27_kernel<<<1, 32, 0, stream>>>(off_b, mod_b, scale27, bias27);

    dim3 cgrid(NPIX / 64, NIMG);
    // conv1 + BN + ReLU  (256 -> 64)
    conv3x3_wmma_kernel<256, 64><<<cgrid, 128, 0, stream>>>(x, wb1T, scale1, bias1, y1, 1);
    // offset (18) + modulation-raw (9) conv  (64 -> 27)
    conv3x3_wmma_kernel<64, 27><<<cgrid, 128, 0, stream>>>(y1, wb2T, scale27, bias27, out27, 0);
    // modulated deformable conv (64 -> 64)
    dcn_wmma_kernel<<<cgrid, 128, 0, stream>>>(y1, out27, wbdT, dcn_b, y2);
    // mask pooling
    pool_kernel<<<dim3(NIMG, 10), 256, 0, stream>>>(y2, masks, numb, denb);
    // FC head -> d_out (inc 20480 floats, then pix 10240 floats)
    head_kernel<<<160, 128, 0, stream>>>(numb, denb, smooth_w, smooth_b, inc_w, pix_w,
                                         (float*)d_out);
}
